// PointNetFeaturePropagation_14817637171237
// MI455X (gfx1250) — matrix operand order repaired and checked by reading, compiled, not run on hardware
//
#include <hip/hip_runtime.h>
#include <hip/hip_bf16.h>

// ---------------------------------------------------------------------------
// PointNet Feature Propagation for MI455X (gfx1250, wave32, WMMA + async LDS)
// B=8, N=8192, S=2048, C=3, D=256, MLP=[256,128]
// ---------------------------------------------------------------------------

typedef __attribute__((ext_vector_type(16))) _Float16 v16h;
typedef __attribute__((ext_vector_type(8)))  _Float16 v8h;
typedef __attribute__((ext_vector_type(8)))  float    v8f;
typedef __attribute__((address_space(3)))    _Float16 lds_f16;

#define Bc 8
#define Nc 8192
#define Sc 2048
#define Dc 256
#define K1 512   // GEMM1 K  (2*D)
#define N1 256   // GEMM1 out cols
#define K2 256   // GEMM2 K
#define N2 128   // GEMM2 out cols
#define Mrows (Bc * Nc)   // 65536

// ---------------------------------------------------------------------------
// Kernel 0: fold BN into scale/bias; transpose+convert weights to f16 [n][k]
// ---------------------------------------------------------------------------
__global__ __launch_bounds__(256) void fp_prep(
    const float* __restrict__ w1, const float* __restrict__ b1,
    const float* __restrict__ g1, const float* __restrict__ be1,
    const float* __restrict__ m1, const float* __restrict__ v1,
    const float* __restrict__ w2, const float* __restrict__ b2,
    const float* __restrict__ g2, const float* __restrict__ be2,
    const float* __restrict__ m2, const float* __restrict__ v2,
    _Float16* __restrict__ wT1, _Float16* __restrict__ wT2,
    float* __restrict__ a1, float* __restrict__ c1,
    float* __restrict__ a2, float* __restrict__ c2)
{
    int i = blockIdx.x * 256 + threadIdx.x;
    if (i < K1 * N1) {                       // wT1[n][k] = w1[k][n]
        int nn = i >> 9, kk = i & 511;
        wT1[i] = (_Float16)w1[(size_t)kk * N1 + nn];
    } else if (i < K1 * N1 + K2 * N2) {      // wT2[n][k] = w2[k][n]
        int t = i - K1 * N1;
        int nn = t >> 8, kk = t & 255;
        wT2[t] = (_Float16)w2[(size_t)kk * N2 + nn];
    } else if (i < K1 * N1 + K2 * N2 + N1) { // BN fold layer 1
        int j = i - (K1 * N1 + K2 * N2);
        float s = g1[j] * rsqrtf(v1[j] + 1e-5f);
        a1[j] = s;
        c1[j] = (b1[j] - m1[j]) * s + be1[j];
    } else if (i < K1 * N1 + K2 * N2 + N1 + N2) { // BN fold layer 2
        int j = i - (K1 * N1 + K2 * N2 + N1);
        float s = g2[j] * rsqrtf(v2[j] + 1e-5f);
        a2[j] = s;
        c2[j] = (b2[j] - m2[j]) * s + be2[j];
    }
}

// ---------------------------------------------------------------------------
// Kernel 1: pack candidates: cand[b][s] = (x, y, z, |p|^2)
// ---------------------------------------------------------------------------
__global__ __launch_bounds__(256) void fp_cand_pack(
    const float* __restrict__ xyz2, float4* __restrict__ cand)
{
    int i = blockIdx.x * 256 + threadIdx.x;      // B*S = 16384
    int b = i >> 11, s = i & (Sc - 1);
    const float* p = xyz2 + (size_t)b * 3 * Sc;
    float x = p[s], y = p[Sc + s], z = p[2 * Sc + s];
    cand[i] = make_float4(x, y, z, __builtin_fmaf(x, x, __builtin_fmaf(y, y, z * z)));
}

// ---------------------------------------------------------------------------
// Kernel 2: tiled transpose points1 [B,D,N] -> X[:, 0:256] (f16, row m = b*N+n)
// ---------------------------------------------------------------------------
__global__ __launch_bounds__(256) void fp_transpose_p1(
    const float* __restrict__ p1, _Float16* __restrict__ X)
{
    __shared__ float tile[32][33];
    const int tx = threadIdx.x, ty = threadIdx.y;
    const int b = blockIdx.z, nB = blockIdx.x * 32, dB = blockIdx.y * 32;
    const float* src = p1 + ((size_t)b * Dc + dB) * Nc + nB;
#pragma unroll
    for (int k = 0; k < 4; ++k)
        tile[ty + 8 * k][tx] = src[(size_t)(ty + 8 * k) * Nc + tx];
    __syncthreads();
    _Float16* dst = X + ((size_t)b * Nc + nB) * K1 + dB;
#pragma unroll
    for (int k = 0; k < 4; ++k)
        dst[(size_t)(ty + 8 * k) * K1 + tx] = (_Float16)tile[tx][ty + 8 * k];
}

// ---------------------------------------------------------------------------
// Kernel 3: tiled transpose points2 [B,D,S] -> f2t [B,S,D] f16
// ---------------------------------------------------------------------------
__global__ __launch_bounds__(256) void fp_transpose_p2(
    const float* __restrict__ p2, _Float16* __restrict__ f2t)
{
    __shared__ float tile[32][33];
    const int tx = threadIdx.x, ty = threadIdx.y;
    const int b = blockIdx.z, sB = blockIdx.x * 32, dB = blockIdx.y * 32;
    const float* src = p2 + ((size_t)b * Dc + dB) * Sc + sB;
#pragma unroll
    for (int k = 0; k < 4; ++k)
        tile[ty + 8 * k][tx] = src[(size_t)(ty + 8 * k) * Sc + tx];
    __syncthreads();
    _Float16* dst = f2t + ((size_t)b * Sc + sB) * Dc + dB;
#pragma unroll
    for (int k = 0; k < 4; ++k)
        dst[(size_t)(ty + 8 * k) * Dc + tx] = (_Float16)tile[tx][ty + 8 * k];
}

// ---------------------------------------------------------------------------
// Kernel 4: 3-NN + inverse-distance interpolation -> X[:, 256:512]
// ---------------------------------------------------------------------------
#define INS3(e_, j_)                                                          \
    do {                                                                      \
        float _e = (e_); int _j = (j_);                                       \
        if (_e < d2[q]) {                                                     \
            if (_e < d1[q]) {                                                 \
                d2[q] = d1[q]; i2[q] = i1[q];                                 \
                if (_e < d0[q]) { d1[q] = d0[q]; i1[q] = i0[q];               \
                                  d0[q] = _e; i0[q] = _j; }                   \
                else            { d1[q] = _e; i1[q] = _j; }                   \
            } else { d2[q] = _e; i2[q] = _j; }                                \
        }                                                                     \
    } while (0)

__global__ __launch_bounds__(256) void fp_knn_interp(
    const float*  __restrict__ xyz1, const float4* __restrict__ cand,
    const _Float16* __restrict__ f2t, _Float16* __restrict__ X)
{
    const int lane = threadIdx.x & 31;
    const int wave = threadIdx.x >> 5;
    const long long q0 = ((long long)blockIdx.x * 8 + wave) * 4; // first query row
    const int b  = (int)(q0 >> 13);                              // /N
    const int n0 = (int)(q0 & (Nc - 1));
    const float* xb = xyz1 + (size_t)b * 3 * Nc;

    float mx[4], my[4], mz[4];
    float d0[4], d1[4], d2[4];
    int   i0[4], i1[4], i2[4];
#pragma unroll
    for (int q = 0; q < 4; ++q) {
        mx[q] = -2.0f * xb[n0 + q];
        my[q] = -2.0f * xb[Nc + n0 + q];
        mz[q] = -2.0f * xb[2 * Nc + n0 + q];
        d0[q] = d1[q] = d2[q] = 3.0e38f;
        i0[q] = i1[q] = i2[q] = 0;
    }

    const float4* cb = cand + (size_t)b * Sc;
    for (int it = 0; it < Sc / 32; ++it) {
        int s = it * 32 + lane;
        float4 c = cb[s];
#pragma unroll
        for (int q = 0; q < 4; ++q) {
            float r = __builtin_fmaf(c.x, mx[q],
                      __builtin_fmaf(c.y, my[q],
                      __builtin_fmaf(c.z, mz[q], c.w)));
            INS3(r, s);
        }
    }

    // cross-lane merge (all lanes converge to the same top-3)
#pragma unroll
    for (int q = 0; q < 4; ++q) {
        for (int off = 16; off > 0; off >>= 1) {
            float e0 = __shfl_xor(d0[q], off, 32);
            float e1 = __shfl_xor(d1[q], off, 32);
            float e2 = __shfl_xor(d2[q], off, 32);
            int   j0 = __shfl_xor(i0[q], off, 32);
            int   j1 = __shfl_xor(i1[q], off, 32);
            int   j2 = __shfl_xor(i2[q], off, 32);
            INS3(e0, j0);
            INS3(e1, j1);
            INS3(e2, j2);
        }
    }

    // weights + coalesced 512B row gathers + f16 store into X[:, 256:512]
#pragma unroll
    for (int q = 0; q < 4; ++q) {
        float qx = -0.5f * mx[q], qy = -0.5f * my[q], qz = -0.5f * mz[q];
        float q2 = qx * qx + qy * qy + qz * qz;
        float t0 = d0[q] + q2, t1 = d1[q] + q2, t2 = d2[q] + q2;
        float w0 = 1.0f / (t0 + 1e-8f);
        float w1 = 1.0f / (t1 + 1e-8f);
        float w2 = 1.0f / (t2 + 1e-8f);
        float inv = 1.0f / (w0 + w1 + w2);
        w0 *= inv; w1 *= inv; w2 *= inv;

        const v8h* f0 = (const v8h*)(f2t + ((size_t)b * Sc + i0[q]) * Dc);
        const v8h* f1 = (const v8h*)(f2t + ((size_t)b * Sc + i1[q]) * Dc);
        const v8h* f2 = (const v8h*)(f2t + ((size_t)b * Sc + i2[q]) * Dc);
        v8h A = f0[lane], Bv = f1[lane], Cv = f2[lane];
        v8h o;
#pragma unroll
        for (int e = 0; e < 8; ++e) {
            float val = w0 * (float)A[e] + w1 * (float)Bv[e] + w2 * (float)Cv[e];
            o[e] = (_Float16)val;
        }
        *((v8h*)(X + (size_t)(q0 + q) * K1 + Dc) + lane) = o;
    }
}

// ---------------------------------------------------------------------------
// GEMM fragment layouts used here (wave32; A and B share one k order):
//   A 16x32 f16: lane L holds row M=L%16; h=L/16: K = 16h + i  (i = 0..15)
//   B 32x16 f16: lane L holds col N=L%16; K = 16h + i  (i = 0..15)
//   C/D 16x16 f32: VGPR v, lane L: M = v + 8h, N = L%16
// W slice (64 cols x K) is staged in LDS with GLOBAL_LOAD_ASYNC_TO_LDS_B128;
// 16B-chunk index XOR-swizzled by (col&7)<<1 to avoid bank conflicts
// (row stride is 0 mod 256B; swizzle keeps the two frag halves adjacent).
// ---------------------------------------------------------------------------
#define SHUF16(lo, hi) __builtin_shufflevector((lo), (hi), 0, 1, 2, 3, 4, 5, 6, 7, 8, 9, 10, 11, 12, 13, 14, 15)

#define ASYNC_LDS_B128(ldsoff, goff, gbase)                                   \
    asm volatile("global_load_async_to_lds_b128 %0, %1, %2"                   \
                 :: "v"(ldsoff), "v"(goff), "s"(gbase) : "memory")
#define WAIT_ASYNC0() asm volatile("s_wait_asynccnt 0x0" ::: "memory")

// GEMM1: Y1 = relu(bn(X @ W1))  — M=65536, K=512, Ncols=256
// grid (M/128, 256/64), block 256 (8 waves); wave = 16 rows x 64 cols
__global__ __launch_bounds__(256) void fp_gemm1(
    const _Float16* __restrict__ X, const _Float16* __restrict__ W,
    const float* __restrict__ a1, const float* __restrict__ c1,
    _Float16* __restrict__ Y)
{
    __shared__ _Float16 wlds[64 * K1];           // 64 KB
    const int lane = threadIdx.x & 31;
    const int wave = threadIdx.x >> 5;
    const int h = lane >> 4, lr = lane & 15;
    const size_t rowBase = (size_t)blockIdx.x * 128 + (size_t)wave * 16;
    const int colBase = blockIdx.y * 64;

    // ---- async-stage the 64-col W slice into LDS (4096 x 16B chunks) ----
    {
        unsigned lbase = (unsigned)(uintptr_t)(lds_f16*)wlds;
        const char* gbase = (const char*)(W + (size_t)colBase * K1);
#pragma unroll
        for (int i = 0; i < 16; ++i) {
            int cl  = i * 256 + (int)threadIdx.x;   // linear chunk id
            int col = cl >> 6, c = cl & 63;         // 64 chunks per 1KB row
            unsigned loff = lbase + (unsigned)((col << 10) + ((c ^ ((col & 7) << 1)) << 4));
            unsigned goff = (unsigned)(cl << 4);
            ASYNC_LDS_B128(loff, goff, gbase);
        }
        WAIT_ASYNC0();
        __syncthreads();
    }

    v8f acc[4];
#pragma unroll
    for (int t = 0; t < 4; ++t) acc[t] = (v8f)(0.0f);

    const _Float16* xr = X + (rowBase + (size_t)lr) * K1;
    const int sw = (lr & 7) << 1;

    for (int k0 = 0; k0 < K1; k0 += 32) {
        __builtin_prefetch(xr + k0 + 128, 0, 3);     // global_prefetch_b8
        v8h alo = *(const v8h*)(xr + k0 + 16 * h);
        v8h ahi = *(const v8h*)(xr + k0 + 16 * h + 8);
        v16h af = SHUF16(alo, ahi);
        const int c0 = (k0 >> 3) + 2 * h;            // even 16B-chunk index
#pragma unroll
        for (int t = 0; t < 4; ++t) {
            const char* bp = (const char*)(wlds + (size_t)(t * 16 + lr) * K1);
            v8h blo = *(const v8h*)(bp + ((c0 ^ sw) << 4));
            v8h bhi = *(const v8h*)(bp + (((c0 + 1) ^ sw) << 4));
            v16h bf = SHUF16(blo, bhi);
            acc[t] = __builtin_amdgcn_wmma_f32_16x16x32_f16(
                false, af, false, bf, (short)0, acc[t], false, false);
        }
    }

#pragma unroll
    for (int t = 0; t < 4; ++t) {
        int col = colBase + t * 16 + lr;
        float s = a1[col], cc = c1[col];
#pragma unroll
        for (int v = 0; v < 8; ++v) {
            float y = fmaxf(acc[t][v] * s + cc, 0.0f);
            Y[(rowBase + (size_t)v + 8 * h) * N1 + col] = (_Float16)y;
        }
    }
}

// GEMM2: out[b][j][n] = relu(bn(Y1 @ W2)) — M=65536, K=256, Ncols=128,
// output transposed to [B,128,N] via padded-LDS staging + float4 stores.
__global__ __launch_bounds__(256) void fp_gemm2(
    const _Float16* __restrict__ Y1, const _Float16* __restrict__ W,
    const float* __restrict__ a2, const float* __restrict__ c2,
    float* __restrict__ out)
{
    __shared__ _Float16 wlds[64 * K2];   // 32 KB
    __shared__ float tile[64 * 132];     // [colLocal][rowLocal], pad 132
    const int lane = threadIdx.x & 31;
    const int wave = threadIdx.x >> 5;
    const int h = lane >> 4, lr = lane & 15;
    const size_t rowBase = (size_t)blockIdx.x * 128 + (size_t)wave * 16;
    const int colBase = blockIdx.y * 64;

    // ---- async-stage the 64-col W slice into LDS (2048 x 16B chunks) ----
    {
        unsigned lbase = (unsigned)(uintptr_t)(lds_f16*)wlds;
        const char* gbase = (const char*)(W + (size_t)colBase * K2);
#pragma unroll
        for (int i = 0; i < 8; ++i) {
            int cl  = i * 256 + (int)threadIdx.x;
            int col = cl >> 5, c = cl & 31;          // 32 chunks per 512B row
            unsigned loff = lbase + (unsigned)((col << 9) + ((c ^ ((col & 7) << 1)) << 4));
            unsigned goff = (unsigned)(cl << 4);
            ASYNC_LDS_B128(loff, goff, gbase);
        }
        WAIT_ASYNC0();
        __syncthreads();
    }

    v8f acc[4];
#pragma unroll
    for (int t = 0; t < 4; ++t) acc[t] = (v8f)(0.0f);

    const _Float16* xr = Y1 + (rowBase + (size_t)lr) * K2;
    const int sw = (lr & 7) << 1;

    for (int k0 = 0; k0 < K2; k0 += 32) {
        __builtin_prefetch(xr + k0 + 128, 0, 3);
        v8h alo = *(const v8h*)(xr + k0 + 16 * h);
        v8h ahi = *(const v8h*)(xr + k0 + 16 * h + 8);
        v16h af = SHUF16(alo, ahi);
        const int c0 = (k0 >> 3) + 2 * h;
#pragma unroll
        for (int t = 0; t < 4; ++t) {
            const char* bp = (const char*)(wlds + (size_t)(t * 16 + lr) * K2);
            v8h blo = *(const v8h*)(bp + ((c0 ^ sw) << 4));
            v8h bhi = *(const v8h*)(bp + (((c0 + 1) ^ sw) << 4));
            v16h bf = SHUF16(blo, bhi);
            acc[t] = __builtin_amdgcn_wmma_f32_16x16x32_f16(
                false, af, false, bf, (short)0, acc[t], false, false);
        }
    }

#pragma unroll
    for (int t = 0; t < 4; ++t) {
        int cl = t * 16 + lr;
        float s = a2[colBase + cl], cc = c2[colBase + cl];
#pragma unroll
        for (int v = 0; v < 8; ++v) {
            float y = fmaxf(acc[t][v] * s + cc, 0.0f);
            tile[cl * 132 + wave * 16 + v + 8 * h] = y;
        }
    }
    __syncthreads();

    // 128 rows per block stay within one batch b (8192/128 = 64 blocks per b)
    const int bb = (int)(((size_t)blockIdx.x * 128) >> 13);
    const int nBase = (int)(((size_t)blockIdx.x * 128) & (Nc - 1));
    float* ob = out + ((size_t)bb * N2 + colBase) * Nc + nBase;
#pragma unroll
    for (int r = 0; r < 8; ++r) {
        int vi = r * 256 + threadIdx.x;     // 0..2047 float4 slots
        int j  = vi >> 5;                   // local col 0..63
        int nq = vi & 31;                   // 4-float chunk within 128 rows
        float4 val = *(const float4*)&tile[j * 132 + nq * 4];
        *(float4*)(ob + (size_t)j * Nc + nq * 4) = val;
    }
}

// ---------------------------------------------------------------------------
// Launcher
// ---------------------------------------------------------------------------
extern "C" void kernel_launch(void* const* d_in, const int* in_sizes, int n_in,
                              void* d_out, int out_size, void* d_ws, size_t ws_size,
                              hipStream_t stream) {
    (void)in_sizes; (void)n_in; (void)out_size; (void)ws_size;
    const float* xyz1    = (const float*)d_in[0];
    const float* xyz2    = (const float*)d_in[1];
    const float* points1 = (const float*)d_in[2];
    const float* points2 = (const float*)d_in[3];
    const float* w1  = (const float*)d_in[4];
    const float* b1  = (const float*)d_in[5];
    const float* g1  = (const float*)d_in[6];
    const float* be1 = (const float*)d_in[7];
    const float* m1  = (const float*)d_in[8];
    const float* v1  = (const float*)d_in[9];
    const float* w2  = (const float*)d_in[10];
    const float* b2  = (const float*)d_in[11];
    const float* g2  = (const float*)d_in[12];
    const float* be2 = (const float*)d_in[13];
    const float* m2  = (const float*)d_in[14];
    const float* v2  = (const float*)d_in[15];

    char* ws = (char*)d_ws;
    size_t off = 0;
    _Float16* X   = (_Float16*)(ws + off); off += (size_t)Mrows * K1 * 2;   // 64 MB
    _Float16* Y1  = (_Float16*)(ws + off); off += (size_t)Mrows * N1 * 2;   // 32 MB
    _Float16* f2t = (_Float16*)(ws + off); off += (size_t)Bc * Sc * Dc * 2; // 8 MB
    float4*   cand = (float4*)(ws + off);  off += (size_t)Bc * Sc * 16;     // 256 KB
    _Float16* wT1 = (_Float16*)(ws + off); off += (size_t)K1 * N1 * 2;      // 256 KB
    _Float16* wT2 = (_Float16*)(ws + off); off += (size_t)K2 * N2 * 2;      // 64 KB
    float* a1 = (float*)(ws + off); off += 1024;
    float* c1 = (float*)(ws + off); off += 1024;
    float* a2 = (float*)(ws + off); off += 1024;
    float* c2 = (float*)(ws + off); off += 1024;

    fp_prep<<<642, 256, 0, stream>>>(w1, b1, g1, be1, m1, v1,
                                     w2, b2, g2, be2, m2, v2,
                                     wT1, wT2, a1, c1, a2, c2);
    fp_cand_pack<<<(Bc * Sc) / 256, 256, 0, stream>>>(xyz2, cand);
    fp_transpose_p2<<<dim3(Sc / 32, Dc / 32, Bc), dim3(32, 8), 0, stream>>>(points2, f2t);
    fp_transpose_p1<<<dim3(Nc / 32, Dc / 32, Bc), dim3(32, 8), 0, stream>>>(points1, X);
    fp_knn_interp<<<Mrows / 32, 256, 0, stream>>>(xyz1, cand, f2t, X);
    fp_gemm1<<<dim3(Mrows / 128, N1 / 64), 256, 0, stream>>>(X, wT1, a1, c1, Y1);
    fp_gemm2<<<dim3(Mrows / 128, N2 / 64), 256, 0, stream>>>(Y1, wT2, a2, c2, (float*)d_out);
}
